// InterlacedSparseSelfAttention_13082470383981
// MI455X (gfx1250) — compile-verified, matrix operand order checked by reading, not run on hardware
//
#include <hip/hip_runtime.h>
#include <hip/hip_bf16.h>

typedef __attribute__((ext_vector_type(16))) __bf16        v16bf;
typedef __attribute__((ext_vector_type(8)))  float         v8f;
typedef __attribute__((ext_vector_type(8)))  unsigned int  v8u;
typedef __attribute__((ext_vector_type(4)))  unsigned int  u32x4;

#define CDIM 256

// ---- helpers -------------------------------------------------------------

__device__ __forceinline__ __bf16 f2bf(float f) {
    unsigned u = __builtin_bit_cast(unsigned, f);
    unsigned r = (u + 0x7FFFu + ((u >> 16) & 1u)) >> 16;   // round-to-nearest-even
    unsigned short s = (unsigned short)r;
    return __builtin_bit_cast(__bf16, s);
}

__device__ __forceinline__ float hred_max16(float v) {
    v = fmaxf(v, __shfl_xor(v, 1));
    v = fmaxf(v, __shfl_xor(v, 2));
    v = fmaxf(v, __shfl_xor(v, 4));
    v = fmaxf(v, __shfl_xor(v, 8));
    return v;   // reduced within each 16-lane half of the wave32
}

__device__ __forceinline__ float hred_sum16(float v) {
    v += __shfl_xor(v, 1);
    v += __shfl_xor(v, 2);
    v += __shfl_xor(v, 4);
    v += __shfl_xor(v, 8);
    return v;
}

__device__ __forceinline__ v16bf make_frag(u32x4 lo, u32x4 hi) {
    v8u r;
    r[0] = lo[0]; r[1] = lo[1]; r[2] = lo[2]; r[3] = lo[3];
    r[4] = hi[0]; r[5] = hi[1]; r[6] = hi[2]; r[7] = hi[3];
    return __builtin_bit_cast(v16bf, r);
}

// A fragment (16x32 bf16, ISA 7.12.2): ap0 already includes (+row*lda + half*8)
__device__ __forceinline__ v16bf loadA(const __bf16* ap0, int ks) {
    u32x4 lo = *(const u32x4*)(ap0 + ks * 32);
    u32x4 hi = *(const u32x4*)(ap0 + ks * 32 + 16);
    return make_frag(lo, hi);
}

// B fragment (32x16 bf16) from an LDS panel of [n][32] rows
__device__ __forceinline__ v16bf ldsBfrag(const __bf16* pan, int nt, int row, int half) {
    const __bf16* p = pan + (nt * 16 + row) * 32 + half * 8;
    return make_frag(*(const u32x4*)p, *(const u32x4*)(p + 16));
}

// Cooperative copy of one [NB x 32] B-slice (B n-major, row stride ldb),
// split into load (global->regs) and store (regs->LDS) for pipelining.
template <int CH>
__device__ __forceinline__ void stage_load(u32x4 c[CH], const __bf16* Bt, int ldb,
                                           int k0, int tid) {
#pragma unroll
    for (int i = 0; i < CH; ++i) {
        int e = (tid * CH + i) * 8;         // bf16 elements into packed slice
        int n = e >> 5, off = e & 31;       // 32 elems per row
        c[i] = *(const u32x4*)(Bt + (size_t)n * ldb + k0 + off);
    }
}

template <int CH>
__device__ __forceinline__ void stage_store(const u32x4 c[CH], __bf16* pan, int tid) {
#pragma unroll
    for (int i = 0; i < CH; ++i)
        *(u32x4*)(pan + (tid * CH + i) * 8) = c[i];
}

// Staged, double-buffered wave-tile GEMM: one wave accumulates a 16 x NB tile of
// A(16 x 32*KSTEPS) @ Bt^T, B panel shared by the whole 128-thread workgroup.
// Inner loop uses a 4-buffer ping-pong so every WMMA's B fragment was ds_loaded
// two WMMAs earlier (nonzero dscnt waits, loads in flight under compute).
template <int NB, int KSTEPS>
__device__ __forceinline__ void staged_mm(const __bf16* ap0,               // per-lane A row ptr
                                          const __bf16* __restrict__ Bt,   // [NB][32*KSTEPS] n-major
                                          int ldb, __bf16* panel,          // [2][NB*32] LDS
                                          int tid, int row, int half, v8f* acc) {
    constexpr int NT = NB / 16;              // even (4 or 16) in all uses
    constexpr int CH = NB / 32;              // 16B chunks per thread per slice
    u32x4 creg[CH];
    stage_load<CH>(creg, Bt, ldb, 0, tid);
    stage_store<CH>(creg, panel, tid);
    v16bf aCur = loadA(ap0, 0);
    __syncthreads();
#pragma unroll
    for (int ks = 0; ks < KSTEPS; ++ks) {
        const __bf16* pan = panel + (ks & 1) * (NB * 32);
        if (ks + 1 < KSTEPS) stage_load<CH>(creg, Bt, ldb, (ks + 1) * 32, tid);   // overlap
        v16bf af = aCur;
        if (ks + 1 < KSTEPS) aCur = loadA(ap0, ks + 1);                            // overlap
        v16bf bA = ldsBfrag(pan, 0, row, half);
        v16bf bB = ldsBfrag(pan, 1, row, half);
#pragma unroll
        for (int nt = 0; nt < NT; nt += 2) {
            v16bf bC = bA, bD = bB;
            if (nt + 2 < NT) bC = ldsBfrag(pan, nt + 2, row, half);   // prefetch
            if (nt + 3 < NT) bD = ldsBfrag(pan, nt + 3, row, half);   // prefetch
            acc[nt]     = __builtin_amdgcn_wmma_f32_16x16x32_bf16(
                false, af, false, bA, (short)0, acc[nt], false, false);
            acc[nt + 1] = __builtin_amdgcn_wmma_f32_16x16x32_bf16(
                false, af, false, bB, (short)0, acc[nt + 1], false, false);
            bA = bC; bB = bD;
        }
        if (ks + 1 < KSTEPS) stage_store<CH>(creg, panel + ((ks + 1) & 1) * (NB * 32), tid);
        __syncthreads();
    }
}

// ---- weight packing: W[k][n] f32 -> Wt[n][k] bf16 (4 matrices) -----------

__global__ void issa_packw(const float* __restrict__ Wq, const float* __restrict__ Wk,
                           const float* __restrict__ Wv, const float* __restrict__ Wo,
                           __bf16* __restrict__ Wt)
{
    int idx = blockIdx.x * blockDim.x + threadIdx.x;      // 4*65536 threads
    int w = idx >> 16, rem = idx & 65535;
    int k = rem >> 8, n = rem & 255;
    const float* W = (w == 0) ? Wq : (w == 1) ? Wk : (w == 2) ? Wv : Wo;
    Wt[(size_t)w * 65536 + (size_t)n * CDIM + k] = f2bf(W[rem]);
}

// ---- gather x[b,c,h,w] -> X1[b*64+ph*8+pw][qh*16+qw][c] bf16 -------------

__global__ void issa_gather1(const float* __restrict__ x, __bf16* __restrict__ X1)
{
    size_t idx = (size_t)blockIdx.x * blockDim.x + threadIdx.x;  // 8*256*128*128
    int ww = (int)(idx & 127);
    int hh = (int)((idx >> 7) & 127);
    int c  = (int)((idx >> 14) & 255);
    int b  = (int)(idx >> 22);
    int qw = ww >> 3, pw = ww & 7, qh = hh >> 3, ph = hh & 7;
    size_t dst = (((size_t)(b * 64 + ph * 8 + pw)) * CDIM + (qh * 16 + qw)) * CDIM + c;
    X1[dst] = f2bf(x[idx]);
}

// ---- batched QKV projection (blockIdx.z selects Wq/Wk/Wv; V stored T) ----

__global__ __launch_bounds__(128)
void issa_qkv(const __bf16* __restrict__ X, const __bf16* __restrict__ Wt3,
              const float* __restrict__ bq, const float* __restrict__ bk,
              const float* __restrict__ bv,
              __bf16* __restrict__ Qo, __bf16* __restrict__ Ko,
              __bf16* __restrict__ Vo, int M)
{
    __shared__ __bf16 panel[2 * 256 * 32];                 // 32 KB
    const int wsel = blockIdx.z;
    const __bf16* Bt   = Wt3 + (size_t)wsel * 65536;
    const float*  bias = (wsel == 0) ? bq : (wsel == 1) ? bk : bv;
    const int batch = blockIdx.y;
    const int tid = threadIdx.x;
    const int wave = tid >> 5, lane = tid & 31;
    const int row = lane & 15, half = lane >> 4;
    const int m0 = (blockIdx.x * 4 + wave) * 16;
    const __bf16* ap0 = X + (size_t)batch * M * CDIM + (size_t)(m0 + row) * CDIM + half * 8;

    v8f zero = {};
    v8f acc[16];
#pragma unroll
    for (int t = 0; t < 16; ++t) acc[t] = zero;
    staged_mm<256, 8>(ap0, Bt, CDIM, panel, tid, row, half, acc);

    if (wsel < 2) {
        __bf16* Out = ((wsel == 0) ? Qo : Ko) + (size_t)batch * M * CDIM;
#pragma unroll
        for (int nt = 0; nt < 16; ++nt) {
            int n = nt * 16 + row; float bvv = bias[n];
#pragma unroll
            for (int r = 0; r < 8; ++r)
                Out[(size_t)(m0 + r + 8 * half) * CDIM + n] = f2bf(acc[nt][r] + bvv);
        }
    } else {  // V transposed: [batch][feat n][token m]
        __bf16* Out = Vo + (size_t)batch * M * CDIM;
#pragma unroll
        for (int nt = 0; nt < 16; ++nt) {
            int n = nt * 16 + row; float bvv = bias[n];
#pragma unroll
            for (int r = 0; r < 8; ++r)
                Out[(size_t)n * M + (m0 + r + 8 * half)] = f2bf(acc[nt][r] + bvv);
        }
    }
}

// ---- attention: S = QK^T * s, softmax, O = P V  (one wave = 16 q-rows) ---

template <int NKV>
__global__ __launch_bounds__(128)
void issa_attn(const __bf16* __restrict__ Q, const __bf16* __restrict__ Km,
               const __bf16* __restrict__ Vt, __bf16* __restrict__ O)
{
    constexpr int NT = NKV / 16;
    __shared__ __bf16 panel[2 * 256 * 32];   // 32 KB, shared by S and PV phases
    __shared__ __bf16 plds[4 * 16 * NKV];    // per-wave P scratch (32 KB / 8 KB)

    const int batch = blockIdx.y;
    const int tid = threadIdx.x;
    const int wave = tid >> 5, lane = tid & 31;
    const int row = lane & 15, half = lane >> 4;
    const int m0 = (blockIdx.x * 4 + wave) * 16;

    const __bf16* Qb = Q  + (size_t)batch * NKV * CDIM;
    const __bf16* Kb = Km + (size_t)batch * NKV * CDIM;
    const __bf16* Vb = Vt + (size_t)batch * CDIM * NKV;

    v8f zero = {};

    // ---- S = Q @ K^T (B = K rows, n-major over tokens, k = features) ----
    v8f s[NT];
#pragma unroll
    for (int t = 0; t < NT; ++t) s[t] = zero;
    const __bf16* qp0 = Qb + (size_t)(m0 + row) * CDIM + half * 8;
    staged_mm<NKV, 8>(qp0, Kb, CDIM, panel, tid, row, half, s);

    // ---- softmax over NKV cols per row (rows split across 16-lane halves)
    const float scale = 0.0625f;     // 1/sqrt(256)
    __bf16* myp = plds + wave * 16 * NKV;
#pragma unroll
    for (int r = 0; r < 8; ++r) {
        float mx = -3.0e38f;
#pragma unroll
        for (int nt = 0; nt < NT; ++nt) mx = fmaxf(mx, s[nt][r]);
        mx = hred_max16(mx);
        float sum = 0.f;
#pragma unroll
        for (int nt = 0; nt < NT; ++nt) {
            float e = __expf((s[nt][r] - mx) * scale);
            s[nt][r] = e; sum += e;
        }
        sum = hred_sum16(sum);
        float rinv = 1.0f / sum;
#pragma unroll
        for (int nt = 0; nt < NT; ++nt)
            myp[(r + 8 * half) * NKV + nt * 16 + row] = f2bf(s[nt][r] * rinv);
    }

    // ---- O = P @ V (A from LDS P scratch, B = transposed V, n = features)
    v8f o[16];
#pragma unroll
    for (int t = 0; t < 16; ++t) o[t] = zero;
    const __bf16* pp0 = myp + (size_t)row * NKV + half * 8;
    staged_mm<256, NKV / 32>(pp0, Vb, NKV, panel, tid, row, half, o);

    __bf16* Ob = O + (size_t)batch * NKV * CDIM;
#pragma unroll
    for (int nt = 0; nt < 16; ++nt)
#pragma unroll
        for (int r = 0; r < 8; ++r)
            Ob[(size_t)(m0 + r + 8 * half) * CDIM + nt * 16 + row] = f2bf(o[nt][r]);
}

// ---- stage-1 out-proj, fused permutation into stage-2 layout -------------

__global__ __launch_bounds__(128)
void issa_proj1(const __bf16* __restrict__ O1, const __bf16* __restrict__ Wot,
                const float* __restrict__ bo, __bf16* __restrict__ X2)
{
    __shared__ __bf16 panel[2 * 256 * 32];
    const int batch = blockIdx.y;           // b*64 + ph*8 + pw
    const int b = batch >> 6, phpw = batch & 63;
    const int tid = threadIdx.x;
    const int wave = tid >> 5, lane = tid & 31;
    const int row = lane & 15, half = lane >> 4;
    const int m0 = (blockIdx.x * 4 + wave) * 16;
    const __bf16* ap0 = O1 + (size_t)batch * CDIM * CDIM + (size_t)(m0 + row) * CDIM + half * 8;

    v8f zero = {};
    v8f acc[16];
#pragma unroll
    for (int t = 0; t < 16; ++t) acc[t] = zero;
    staged_mm<256, 8>(ap0, Wot, CDIM, panel, tid, row, half, acc);

#pragma unroll
    for (int nt = 0; nt < 16; ++nt) {
        int n = nt * 16 + row; float bvv = bo[n];
#pragma unroll
        for (int r = 0; r < 8; ++r) {
            int m = m0 + r + 8 * half;      // qh*16+qw
            size_t dst = (((size_t)(b * 256 + m)) * 64 + phpw) * CDIM + n;
            X2[dst] = f2bf(acc[nt][r] + bvv);
        }
    }
}

// ---- stage-2 out-proj, fused scatter to [b,c,h,w] fp32 -------------------

__global__ __launch_bounds__(128)
void issa_proj2(const __bf16* __restrict__ O2, const __bf16* __restrict__ Wot,
                const float* __restrict__ bo, float* __restrict__ Y)
{
    __shared__ __bf16 panel[2 * 256 * 32];
    const int batch = blockIdx.y;           // b*256 + qh*16 + qw
    const int b = batch >> 8, qhqw = batch & 255;
    const int qh = qhqw >> 4, qw = qhqw & 15;
    const int tid = threadIdx.x;
    const int wave = tid >> 5, lane = tid & 31;
    const int row = lane & 15, half = lane >> 4;
    const int m0 = wave * 16;               // M = 64, grid.x = 1
    const __bf16* ap0 = O2 + (size_t)batch * 64 * CDIM + (size_t)(m0 + row) * CDIM + half * 8;

    v8f zero = {};
    v8f acc[16];
#pragma unroll
    for (int t = 0; t < 16; ++t) acc[t] = zero;
    staged_mm<256, 8>(ap0, Wot, CDIM, panel, tid, row, half, acc);

#pragma unroll
    for (int nt = 0; nt < 16; ++nt) {
        int n = nt * 16 + row; float bvv = bo[n];
#pragma unroll
        for (int r = 0; r < 8; ++r) {
            int m = m0 + r + 8 * half;      // ph*8+pw
            int ph = m >> 3, pw = m & 7;
            size_t dst = (((size_t)b * CDIM + n) * 128 + (qh * 8 + ph)) * 128 + (qw * 8 + pw);
            Y[dst] = acc[nt][r] + bvv;
        }
    }
}

// ---- host side -----------------------------------------------------------

extern "C" void kernel_launch(void* const* d_in, const int* in_sizes, int n_in,
                              void* d_out, int out_size, void* d_ws, size_t ws_size,
                              hipStream_t stream)
{
    const float* x  = (const float*)d_in[0];
    const float* Wq = (const float*)d_in[1];
    const float* bq = (const float*)d_in[2];
    const float* Wk = (const float*)d_in[3];
    const float* bk = (const float*)d_in[4];
    const float* Wv = (const float*)d_in[5];
    const float* bv = (const float*)d_in[6];
    const float* Wo = (const float*)d_in[7];
    const float* bo = (const float*)d_in[8];

    char* ws = (char*)d_ws;
    const size_t MB64 = 67108864ull;                 // 512*256*256*2 == 2048*64*256*2
    __bf16* Wt  = (__bf16*)(ws);                     // 4 x 256x256 bf16, transposed
    __bf16* X1  = (__bf16*)(ws + (1ull << 20));
    __bf16* Qb  = (__bf16*)(ws + (1ull << 20) + 1 * MB64);
    __bf16* Kb  = (__bf16*)(ws + (1ull << 20) + 2 * MB64);
    __bf16* Vb  = (__bf16*)(ws + (1ull << 20) + 3 * MB64);   // transposed V
    __bf16* Ob  = (__bf16*)(ws + (1ull << 20) + 4 * MB64);
    __bf16* X2  = (__bf16*)(ws + (1ull << 20) + 5 * MB64);

    // weights -> bf16 transposed
    issa_packw<<<1024, 256, 0, stream>>>(Wq, Wk, Wv, Wo, Wt);
    // x -> stage-1 token layout
    issa_gather1<<<131072, 256, 0, stream>>>(x, X1);

    // stage 1: 512 batches, N=256
    issa_qkv<<<dim3(4, 512, 3), 128, 0, stream>>>(X1, Wt, bq, bk, bv, Qb, Kb, Vb, 256);
    issa_attn<256><<<dim3(4, 512), 128, 0, stream>>>(Qb, Kb, Vb, Ob);
    issa_proj1<<<dim3(4, 512), 128, 0, stream>>>(Ob, Wt + 3 * 65536, bo, X2);

    // stage 2: 2048 batches, N=64 (reuse Q/K/V/O buffers)
    issa_qkv<<<dim3(1, 2048, 3), 128, 0, stream>>>(X2, Wt, bq, bk, bv, Qb, Kb, Vb, 64);
    issa_attn<64><<<dim3(1, 2048), 128, 0, stream>>>(Qb, Kb, Vb, Ob);
    issa_proj2<<<dim3(1, 2048), 128, 0, stream>>>(Ob, Wt + 3 * 65536, bo, (float*)d_out);
}
